// RingConv2dCUDA_61856118997344
// MI455X (gfx1250) — matrix operand order repaired and checked by reading, not compiled, not run on hardware
//
#include <hip/hip_runtime.h>
#include <stdint.h>

// ---------------------------------------------------------------------------
// Ring conv (cos/sin decomposition) as implicit-GEMM on CDNA5 WMMA.
//   out[b,o,oh,ow] = sum_{t,c,kh,kw} trig_t(xpad[b,c,oh+kh,ow+kw]) * trig_t(w[o,c,kh,kw])
// K = 2*9*64 = 1152, N = 128 out channels, M = 32*64*64 pixels.
// v_wmma_f32_16x16x32_f16 (wave32) + async global->LDS double-buffered staging.
// ---------------------------------------------------------------------------

typedef __attribute__((ext_vector_type(16))) _Float16 v16h;
typedef __attribute__((ext_vector_type(8)))  _Float16 v8h;
typedef __attribute__((ext_vector_type(8)))  float    v8f;
typedef __attribute__((ext_vector_type(4)))  float    v4f;

#define B_   32
#define C_   64
#define H_   64
#define W_   64
#define O_   128
#define HP   66
#define WP   66
#define KTOT 1152                 // 2 * 9 * 64
static constexpr size_t TRIG_ELEMS = (size_t)B_ * HP * WP * C_;   // per trig plane

__device__ __forceinline__ v16h cat8(v8h lo, v8h hi) {
    v16h r;
#pragma unroll
    for (int i = 0; i < 8; ++i) { r[i] = lo[i]; r[i + 8] = hi[i]; }
    return r;
}

// --- CDNA5 async global->LDS copy (VGLOBAL GVS form), tracked by ASYNCcnt ---
__device__ __forceinline__ void async_ld_b128(uint32_t lds_byte, uint32_t voff_byte,
                                              const void* sbase) {
    asm volatile("global_load_async_to_lds_b128 %0, %1, %2"
                 :: "v"(lds_byte), "v"(voff_byte), "s"(sbase)
                 : "memory");
}
__device__ __forceinline__ void wait_async_le4() {
    asm volatile("s_wait_asynccnt 0x4" ::: "memory");
}
__device__ __forceinline__ void wait_async_0() {
    asm volatile("s_wait_asynccnt 0x0" ::: "memory");
}

// ---------------------------------------------------------------------------
// Stage 1: padded NHWC cos/sin planes in f16. Layout [b][h(66)][w(66)][c(64)].
// Pad positions: cos=1, sin=0 (zero-padded angles of the reference).
// ---------------------------------------------------------------------------
__global__ void ringconv_prep_x(const float* __restrict__ x,
                                _Float16* __restrict__ cosx,
                                _Float16* __restrict__ sinx) {
    size_t idx = (size_t)blockIdx.x * blockDim.x + threadIdx.x;
    if (idx >= TRIG_ELEMS) return;
    int c = (int)(idx & 63);
    size_t t = idx >> 6;
    int w = (int)(t % WP); t /= WP;
    int h = (int)(t % HP);
    int b = (int)(t / HP);
    float cv = 1.0f, sv = 0.0f;
    if (h >= 1 && h <= H_ && w >= 1 && w <= W_) {
        float a = x[(((size_t)b * C_ + c) * H_ + (h - 1)) * W_ + (w - 1)];
        __sincosf(a, &sv, &cv);
    }
    cosx[idx] = (_Float16)cv;
    sinx[idx] = (_Float16)sv;
}

// ---------------------------------------------------------------------------
// Stage 2: weights -> f16 transposed [n][k], k = (t*9+kh*3+kw)*64 + c.
// ---------------------------------------------------------------------------
__global__ void ringconv_prep_w(const float* __restrict__ wgt,
                                _Float16* __restrict__ wT) {
    int idx = blockIdx.x * blockDim.x + threadIdx.x;   // n*KTOT + k
    if (idx >= O_ * KTOT) return;
    int k = idx % KTOT;
    int n = idx / KTOT;
    int c  = k & 63;
    int g  = k >> 6;            // t*9 + kh*3 + kw
    int tt = g / 9;
    int r  = g - tt * 9;
    int kh = r / 3, kw = r - (r / 3) * 3;
    float a = wgt[(((size_t)n * C_ + c) * 3 + kh) * 3 + kw];
    float s, cv;
    __sincosf(a, &s, &cv);
    wT[idx] = (_Float16)(tt ? s : cv);
}

// ---------------------------------------------------------------------------
// Stage 3: implicit-GEMM conv.
//   Block (256 thr = 8 wave32) -> tile M=128 pixels (2 output rows) x N=128.
//   Waves: 4 along M x 2 along N; each wave owns 32x64 -> 8 accumulators.
//   K loop: 36 steps of BK=32, double-buffered async LDS staging.
// ---------------------------------------------------------------------------
__global__ __launch_bounds__(256)
void ringconv_wmma_f16(const _Float16* __restrict__ cosx,
                       const _Float16* __restrict__ sinx,
                       const _Float16* __restrict__ wT,
                       float* __restrict__ out) {
    // [buf][ A: 128x32 halves (8KB) | B: 128x32 halves (8KB) ]  -> 32 KB total
    __shared__ __align__(16) _Float16 smem[2][8192];

    const int tid   = threadIdx.x;
    const int lane  = tid & 31;
    const int wv    = tid >> 5;
    const int waveM = wv & 3;           // M groups of 32 rows
    const int waveN = wv >> 2;          // N groups of 64 channels
    const int l15   = lane & 15;
    const int lhi   = lane >> 4;

    const int b   = blockIdx.x >> 5;          // 1024 blocks = 32 * 32
    const int oh0 = (blockIdx.x & 31) << 1;   // two output rows per block

    const uint32_t ldsBase = (uint32_t)(uintptr_t)(&smem[0][0]);

    // Loop-invariant per-thread staging addresses (byte offsets).
    uint32_t aOff[2], bOff[2], ldsA[2], ldsB[2];
#pragma unroll
    for (int i = 0; i < 2; ++i) {
        int idx = tid + (i << 8);           // 512 16B-chunks per tile
        int row = idx >> 2, chk = idx & 3;  // A: row = pixel 0..127
        int rhi = row >> 6, oww = row & 63;
        aOff[i] = (uint32_t)((((uint32_t)b * HP + (uint32_t)(oh0 + rhi)) * WP + oww) * C_
                             + (chk << 3)) * 2u;
        ldsA[i] = ldsBase + (uint32_t)((row * 32 + (chk << 3)) * 2);
        int n = idx >> 2, ch = idx & 3;     // B: n = out channel 0..127
        bOff[i] = (uint32_t)(n * KTOT + (ch << 3)) * 2u;
        ldsB[i] = ldsBase + (uint32_t)(4096 * 2 + (n * 32 + (ch << 3)) * 2);
    }

    // Issue one K-step's 4 async b128 loads into buffer (kb & 1).
    auto stage = [&](int kb) {
        int g  = kb >> 1;
        int c0 = (kb & 1) << 5;
        int tt = (g >= 9);
        int r  = g - tt * 9;
        int kh = r / 3, kw = r - (r / 3) * 3;
        const char* sa = (const char*)(tt ? sinx : cosx)
                       + (size_t)((kh * WP + kw) * C_ + c0) * 2;   // uniform -> SGPR
        const char* sb = (const char*)wT + (size_t)kb * 64;        // kb*32 halves
        uint32_t bo = (uint32_t)(kb & 1) * (8192u * 2u);
        async_ld_b128(ldsA[0] + bo, aOff[0], sa);
        async_ld_b128(ldsA[1] + bo, aOff[1], sa);
        async_ld_b128(ldsB[0] + bo, bOff[0], sb);
        async_ld_b128(ldsB[1] + bo, bOff[1], sb);
    };

    v8f acc[2][4] = {};

    stage(0);

    for (int kb = 0; kb < 36; ++kb) {
        if (kb < 35) { stage(kb + 1); wait_async_le4(); }  // retire stage kb
        else         { wait_async_0(); }
        __syncthreads();                                   // stage kb visible block-wide

        const _Float16* Ab = &smem[kb & 1][0];
        const _Float16* Bb = &smem[kb & 1][4096];

        // A fragments: lanes 0-15 hold K{0..7,16..23}, lanes 16-31 K{8..15,24..31}
        v16h af[2];
#pragma unroll
        for (int mi = 0; mi < 2; ++mi) {
            int mrow = waveM * 32 + mi * 16 + l15;
            int ksub = lhi << 3;
            v8h alo = *(const v8h*)(Ab + mrow * 32 + ksub);
            v8h ahi = *(const v8h*)(Ab + mrow * 32 + ksub + 16);
            af[mi] = cat8(alo, ahi);
        }
#pragma unroll
        for (int nt = 0; nt < 4; ++nt) {
            int ncol = waveN * 64 + nt * 16 + l15;   // B column per lane
            int koff = lhi << 3;                     // lanes 16-31: K{8..15,24..31}
            v8h blo = *(const v8h*)(Bb + ncol * 32 + koff);
            v8h bhi = *(const v8h*)(Bb + ncol * 32 + koff + 16);
            v16h bf = cat8(blo, bhi);
#pragma unroll
            for (int mi = 0; mi < 2; ++mi)
                acc[mi][nt] = __builtin_amdgcn_wmma_f32_16x16x32_f16(
                    false, af[mi], false, bf, (short)0, acc[mi][nt], false, false);
        }
        __syncthreads();                             // buffer reusable for kb+2
    }

    // Epilogue: C/D layout (N = lane&15 across lanes, M = (lane>>4)*8 + vgpr).
    // Output is write-once (67 MB): nontemporal stores keep the reused trig
    // planes + weights (~36 MB) resident in the 192 MB L2.
#pragma unroll
    for (int mi = 0; mi < 2; ++mi) {
        int rowb = waveM * 32 + mi * 16 + (lhi << 3);   // pixel row base, 8-aligned
        int ohh  = oh0 + (rowb >> 6);
        int owb  = rowb & 63;
#pragma unroll
        for (int nt = 0; nt < 4; ++nt) {
            int och = waveN * 64 + nt * 16 + l15;
            float* op = out + (((size_t)b * O_ + och) * H_ + ohh) * W_ + owb;
            v4f lo = { acc[mi][nt][0], acc[mi][nt][1], acc[mi][nt][2], acc[mi][nt][3] };
            v4f hi = { acc[mi][nt][4], acc[mi][nt][5], acc[mi][nt][6], acc[mi][nt][7] };
            __builtin_nontemporal_store(lo, (v4f*)op);
            __builtin_nontemporal_store(hi, (v4f*)(op + 4));
        }
    }
}

// ---------------------------------------------------------------------------
extern "C" void kernel_launch(void* const* d_in, const int* in_sizes, int n_in,
                              void* d_out, int out_size, void* d_ws, size_t ws_size,
                              hipStream_t stream) {
    const float* x   = (const float*)d_in[0];   // (32,64,64,64) NCHW angles
    const float* wgt = (const float*)d_in[1];   // (128,64,3,3) OIHW angles
    float* out = (float*)d_out;                 // (32,128,64,64) NCHW

    // workspace (f16): cos plane | sin plane | transposed weights  (~36 MB)
    _Float16* cosx = (_Float16*)d_ws;
    _Float16* sinx = cosx + TRIG_ELEMS;
    _Float16* wT   = sinx + TRIG_ELEMS;

    {
        int nblk = (int)((TRIG_ELEMS + 255) / 256);
        ringconv_prep_x<<<nblk, 256, 0, stream>>>(x, cosx, sinx);
    }
    {
        int nblk = (O_ * KTOT + 255) / 256;
        ringconv_prep_w<<<nblk, 256, 0, stream>>>(wgt, wT);
    }
    {
        ringconv_wmma_f16<<<B_ * H_ / 2, 256, 0, stream>>>(cosx, sinx, wT, out);
    }
}
